// GraphFusion_86337432584668
// MI455X (gfx1250) — matrix operand order repaired and checked by reading, not compiled, not run on hardware
//
#include <hip/hip_runtime.h>
#include <math.h>

// ---------------------------------------------------------------------------
// Types for CDNA5 WMMA (gfx1250, wave32)
// ---------------------------------------------------------------------------
typedef __bf16 v16bf __attribute__((ext_vector_type(16)));
typedef float  v8f   __attribute__((ext_vector_type(8)));

__device__ __forceinline__ __bf16 to_bf16(float f) { return (__bf16)f; }

union Frag { v16bf v; uint4 q[2]; };

// Problem constants (match reference setup_inputs)
#define NUMS   3072
#define NN     (2 * NUMS)   // 6144
#define DD     256
#define PP     4
#define KC     (PP * DD)    // 1024 concatenated-K for the gram GEMM
#define NHID   256
#define OFEAT  128
#define EPSV   1e-12f
#define BK     64           // K-tile (two 16x16x32 steps per stage)

// Epilogue modes
#define EPI_NONE   0
#define EPI_GRAM   1
#define EPI_H      2
#define EPI_OUT    3
#define EPI_NONE_T 4   // bf16 store transposed: C[n*ldc + m]

// ---------------------------------------------------------------------------
// CDNA5 async global->LDS DMA (ASYNCcnt-tracked, no VGPR round trip)
// ---------------------------------------------------------------------------
__device__ __forceinline__ void async_load_b128(unsigned lds_off, const void* gptr) {
    asm volatile("global_load_async_to_lds_b128 %0, %1, off"
                 :: "v"(lds_off), "v"(gptr)
                 : "memory");
}
__device__ __forceinline__ void wait_async0() {
    asm volatile("s_wait_asynccnt 0x0" ::: "memory");
}

// ---------------------------------------------------------------------------
// 128x128-tile bf16 WMMA GEMM:  C[m,n] = alpha * sum_k A[m,k] * B[n,k]
// (B always stored row-major [N, ldb] -> contiguous K, async-DMA friendly)
// 256 threads = 8 waves in 4(M) x 2(N); each wave computes 32x64.
// Double-buffered LDS; tiles staged with global_load_async_to_lds_b128.
// ---------------------------------------------------------------------------
template <int EPI>
__global__ __launch_bounds__(256) void gemm_wmma(
    const __bf16* __restrict__ A, int lda,
    const __bf16* __restrict__ B, int ldb,
    void* __restrict__ C, int ldc,
    int Ksize, float alpha,
    const float* __restrict__ invrow,   // EPI_H / EPI_OUT: 1/rowsum per m
    float* __restrict__ rowsumOut,      // EPI_GRAM: accumulated row sums
    const float* __restrict__ bias)     // EPI_H / EPI_OUT: per-n bias
{
    __shared__ __bf16 lA[2][128 * BK];
    __shared__ __bf16 lB[2][128 * BK];

    const int tid   = threadIdx.x;
    const int lane  = tid & 31;
    const int wid   = tid >> 5;
    const int waveM = wid & 3;        // 0..3  -> 32-row slabs
    const int waveN = wid >> 2;       // 0..1  -> 64-col slabs
    const int half  = lane >> 4;      // 0/1 lane half
    const int l15   = lane & 15;

    const size_t mT = (size_t)blockIdx.y * 128;
    const size_t nT = (size_t)blockIdx.x * 128;
    const __bf16* Abase = A + mT * (size_t)lda;
    const __bf16* Bbase = B + nT * (size_t)ldb;

    const unsigned ldsA0 = (unsigned)(uintptr_t)(&lA[0][0]);
    const unsigned ldsA1 = (unsigned)(uintptr_t)(&lA[1][0]);
    const unsigned ldsB0 = (unsigned)(uintptr_t)(&lB[0][0]);
    const unsigned ldsB1 = (unsigned)(uintptr_t)(&lB[1][0]);

    // Stage one 128xBK tile of A and B into LDS buffer `buf` at k-offset kt.
    auto stage = [&](int kt, unsigned aBase, unsigned bBase) {
        #pragma unroll
        for (int c = tid; c < 1024; c += 256) {       // 128 rows x 8 chunks(8 bf16)
            const int row = c >> 3;
            const int c8  = (c & 7) << 3;
            const unsigned loff = (unsigned)(row * BK + c8) * 2u;
            async_load_b128(aBase + loff, Abase + (size_t)row * lda + kt + c8);
            async_load_b128(bBase + loff, Bbase + (size_t)row * ldb + kt + c8);
        }
    };

    v8f acc[2][4] = {};

    const int nTiles = Ksize / BK;
    stage(0, ldsA0, ldsB0);
    wait_async0();
    __syncthreads();

    int buf = 0;
    for (int t = 0; t < nTiles; ++t) {
        if (t + 1 < nTiles)
            stage((t + 1) * BK, buf ? ldsA0 : ldsA1, buf ? ldsB0 : ldsB1);

        const __bf16* cA = &lA[buf][0];
        const __bf16* cB = &lB[buf][0];
        #pragma unroll
        for (int s = 0; s < BK; s += 32) {
            Frag af[2];
            #pragma unroll
            for (int i = 0; i < 2; ++i) {            // A: m = l15, K split per half
                const int r  = waveM * 32 + i * 16 + l15;
                const int k0 = s + half * 8;
                af[i].q[0] = *reinterpret_cast<const uint4*>(&cA[r * BK + k0]);
                af[i].q[1] = *reinterpret_cast<const uint4*>(&cA[r * BK + k0 + 16]);
            }
            Frag bfv[4];
            #pragma unroll
            for (int j = 0; j < 4; ++j) {            // B: n = l15, K split per half
                const int r  = waveN * 64 + j * 16 + l15;
                const int k0 = s + half * 8;
                bfv[j].q[0] = *reinterpret_cast<const uint4*>(&cB[r * BK + k0]);
                bfv[j].q[1] = *reinterpret_cast<const uint4*>(&cB[r * BK + k0 + 16]);
            }
            #pragma unroll
            for (int i = 0; i < 2; ++i)
                #pragma unroll
                for (int j = 0; j < 4; ++j)
                    acc[i][j] = __builtin_amdgcn_wmma_f32_16x16x32_bf16(
                        false, af[i].v, false, bfv[j].v, (short)0, acc[i][j],
                        false, false);
        }

        wait_async0();       // our share of next-tile DMA has landed
        __syncthreads();     // all waves done reading `buf` and staging `buf^1`
        buf ^= 1;
    }

    // ---- epilogue ----
    #pragma unroll
    for (int i = 0; i < 2; ++i) {
        #pragma unroll
        for (int j = 0; j < 4; ++j) {
            #pragma unroll
            for (int r = 0; r < 8; ++r) {
                const size_t m = mT + waveM * 32 + i * 16 + r + (half << 3);
                const size_t n = nT + waveN * 64 + j * 16 + l15;
                float v = alpha * acc[i][j][r];
                if (EPI == EPI_GRAM) {
                    v = v > 0.0f ? v : 0.0f;
                    ((__bf16*)C)[m * ldc + n] = to_bf16(v);
                    float s = v;                      // row-sum over the 16 n-lanes
                    s += __shfl_xor(s, 1, 32);
                    s += __shfl_xor(s, 2, 32);
                    s += __shfl_xor(s, 4, 32);
                    s += __shfl_xor(s, 8, 32);
                    if (l15 == 0) atomicAdd(&rowsumOut[m], s);
                } else if (EPI == EPI_NONE) {
                    ((__bf16*)C)[m * ldc + n] = to_bf16(v);
                } else if (EPI == EPI_NONE_T) {
                    ((__bf16*)C)[n * ldc + m] = to_bf16(v);
                } else if (EPI == EPI_H) {
                    v = v * invrow[m] + bias[n];
                    v = fmaxf(v, 0.0f);
                    ((__bf16*)C)[m * ldc + n] = to_bf16(v);
                } else { // EPI_OUT
                    v = v * invrow[m] + bias[n];
                    ((float*)C)[m * ldc + n] = v;
                }
            }
        }
    }
}

// ---------------------------------------------------------------------------
// ctx prep: emb = concat(stru, mm); per perspective p reweight + L2 normalize.
// One wave per node row; lane handles 8 of 256 elements.
// ---------------------------------------------------------------------------
__global__ __launch_bounds__(32) void prep_ctx(
    const float* __restrict__ stru, const float* __restrict__ mm,
    const float* __restrict__ glw,
    __bf16* __restrict__ embb, __bf16* __restrict__ ctx)
{
    const int n = blockIdx.x;
    const int lane = threadIdx.x;
    const float* src = (n < NUMS) ? (stru + (size_t)n * DD)
                                  : (mm + (size_t)(n - NUMS) * DD);
    float e[8];
    #pragma unroll
    for (int j = 0; j < 8; ++j) {
        e[j] = src[lane + j * 32];
        embb[(size_t)n * DD + lane + j * 32] = to_bf16(e[j]);
    }
    #pragma unroll
    for (int p = 0; p < PP; ++p) {
        float w[8], ss = 0.0f;
        #pragma unroll
        for (int j = 0; j < 8; ++j) {
            w[j] = e[j] * glw[p * DD + lane + j * 32];
            ss += w[j] * w[j];
        }
        ss += __shfl_xor(ss, 16, 32);
        ss += __shfl_xor(ss, 8, 32);
        ss += __shfl_xor(ss, 4, 32);
        ss += __shfl_xor(ss, 2, 32);
        ss += __shfl_xor(ss, 1, 32);
        const float inv = 1.0f / fmaxf(sqrtf(ss), EPSV);
        #pragma unroll
        for (int j = 0; j < 8; ++j)
            ctx[(size_t)n * KC + p * DD + lane + j * 32] = to_bf16(w[j] * inv);
    }
}

// f32 [R][C] -> bf16 transposed [C][R]
__global__ void cvt_bf16_T(const float* __restrict__ in, __bf16* __restrict__ out,
                           int R, int Ccols) {
    int i = blockIdx.x * blockDim.x + threadIdx.x;
    if (i < R * Ccols) {
        int r = i / Ccols, c = i % Ccols;
        out[(size_t)c * R + r] = to_bf16(in[i]);
    }
}

__global__ void zero_f32(float* p, int n) {
    int i = blockIdx.x * blockDim.x + threadIdx.x;
    if (i < n) p[i] = 0.0f;
}

__global__ void invrow_k(float* p, int n) {
    int i = blockIdx.x * blockDim.x + threadIdx.x;
    if (i < n) p[i] = 1.0f / fmaxf(p[i], EPSV);
}

// log_softmax over rows of OFEAT=128: one wave per row, 4 values per lane.
__global__ __launch_bounds__(32) void logsoftmax_k(
    const float* __restrict__ O, float* __restrict__ out)
{
    const int row = blockIdx.x;
    const int lane = threadIdx.x;
    const float* o = O + (size_t)row * OFEAT;
    float v[4], mx = -INFINITY;
    #pragma unroll
    for (int j = 0; j < 4; ++j) { v[j] = o[lane + j * 32]; mx = fmaxf(mx, v[j]); }
    #pragma unroll
    for (int m = 16; m >= 1; m >>= 1) mx = fmaxf(mx, __shfl_xor(mx, m, 32));
    float se = 0.0f;
    #pragma unroll
    for (int j = 0; j < 4; ++j) se += __expf(v[j] - mx);
    #pragma unroll
    for (int m = 16; m >= 1; m >>= 1) se += __shfl_xor(se, m, 32);
    const float lse = mx + __logf(se);
    #pragma unroll
    for (int j = 0; j < 4; ++j) out[(size_t)row * OFEAT + lane + j * 32] = v[j] - lse;
}

// ---------------------------------------------------------------------------
// Host orchestration
// ---------------------------------------------------------------------------
static inline size_t alignUp(size_t x, size_t a) { return (x + a - 1) & ~(a - 1); }

extern "C" void kernel_launch(void* const* d_in, const int* in_sizes, int n_in,
                              void* d_out, int out_size, void* d_ws, size_t ws_size,
                              hipStream_t stream) {
    const float* stru = (const float*)d_in[0];
    const float* mm   = (const float*)d_in[1];
    const float* glw  = (const float*)d_in[2];
    const float* w0   = (const float*)d_in[3];
    const float* b0   = (const float*)d_in[4];
    const float* w1   = (const float*)d_in[5];
    const float* b1   = (const float*)d_in[6];
    (void)in_sizes; (void)n_in; (void)out_size; (void)ws_size;

    // workspace carve-up
    char* ws = (char*)d_ws;
    size_t off = 0;
    auto take = [&](size_t bytes) { char* p = ws + off; off = alignUp(off + bytes, 256); return p; };
    __bf16* ctx  = (__bf16*)take((size_t)NN * KC * 2);        // [N, P*d]
    __bf16* embb = (__bf16*)take((size_t)NN * DD * 2);        // [N, d]
    __bf16* S    = (__bf16*)take((size_t)NN * NN * 2);        // masked gram, bf16
    __bf16* X0T  = (__bf16*)take((size_t)NHID * NN * 2);      // (emb @ w0)^T
    __bf16* H    = (__bf16*)take((size_t)NN * NHID * 2);      // hidden, row-major
    __bf16* GT   = (__bf16*)take((size_t)OFEAT * NN * 2);     // (H @ w1)^T
    float*  O    = (float*) take((size_t)NN * OFEAT * 4);     // pre-softmax
    __bf16* w0bT = (__bf16*)take((size_t)NHID * DD * 2);      // w0^T  [NHID][DD]
    __bf16* w1bT = (__bf16*)take((size_t)OFEAT * NHID * 2);   // w1^T  [OFEAT][NHID]
    float*  rsum = (float*) take((size_t)NN * 4);             // rowsum -> invrow

    // 0) zero rowsum accumulator
    zero_f32<<<(NN + 255) / 256, 256, 0, stream>>>(rsum, NN);
    // 1) ctx prep (+ bf16 emb)
    prep_ctx<<<NN, 32, 0, stream>>>(stru, mm, glw, embb, ctx);
    // 2) bf16 transposed weights
    cvt_bf16_T<<<(DD * NHID + 255) / 256, 256, 0, stream>>>(w0, w0bT, DD, NHID);
    cvt_bf16_T<<<(NHID * OFEAT + 255) / 256, 256, 0, stream>>>(w1, w1bT, NHID, OFEAT);
    // 3) masked gram: S = relu((1/P) * ctx @ ctx^T), rowsums accumulated
    gemm_wmma<EPI_GRAM><<<dim3(NN / 128, NN / 128), 256, 0, stream>>>(
        ctx, KC, ctx, KC, (void*)S, NN, KC, 1.0f / PP, nullptr, rsum, nullptr);
    // 4) invert rowsums (in place)
    invrow_k<<<(NN + 255) / 256, 256, 0, stream>>>(rsum, NN);
    // 5) X0^T = (emb @ w0)^T            (B = w0^T, transposed store)
    gemm_wmma<EPI_NONE_T><<<dim3(NHID / 128, NN / 128), 256, 0, stream>>>(
        embb, DD, w0bT, DD, (void*)X0T, NN, DD, 1.0f, nullptr, nullptr, nullptr);
    // 6) H = relu(diag(invrow) * (S @ X0) + b0)      (B = X0^T)
    gemm_wmma<EPI_H><<<dim3(NHID / 128, NN / 128), 256, 0, stream>>>(
        S, NN, X0T, NN, (void*)H, NHID, NN, 1.0f, rsum, nullptr, b0);
    // 7) G^T = (H @ w1)^T               (B = w1^T, transposed store)
    gemm_wmma<EPI_NONE_T><<<dim3(OFEAT / 128, NN / 128), 256, 0, stream>>>(
        H, NHID, w1bT, NHID, (void*)GT, NN, NHID, 1.0f, nullptr, nullptr, nullptr);
    // 8) O = diag(invrow) * (S @ G) + b1             (B = G^T)
    gemm_wmma<EPI_OUT><<<dim3(OFEAT / 128, NN / 128), 256, 0, stream>>>(
        S, NN, GT, NN, (void*)O, OFEAT, NN, 1.0f, rsum, nullptr, b1);
    // 9) log_softmax rows -> d_out
    logsoftmax_k<<<NN, 32, 0, stream>>>(O, (float*)d_out);
}
